// DEC_90125593739499
// MI455X (gfx1250) — matrix operand order repaired and checked by reading, not compiled, not run on hardware
//
#include <hip/hip_runtime.h>

// Problem constants (from reference)
#define LAYERS 2
#define BATCH  32
#define TSTEPS 128
#define HID    128
#define FIN    3
#define DDELAY 10
#define GATES  384      // 3*HID
#define ROWS   64       // LAYERS*BATCH
#define NTHR   1024     // 32 wave32 waves per workgroup

typedef __attribute__((ext_vector_type(16))) _Float16 v16h;
typedef __attribute__((ext_vector_type(8)))  float    v8f;

// Fast hardware tanh (V_TANH_F32 on gfx1250) with safe fallback.
#if defined(__has_builtin)
#if __has_builtin(__builtin_amdgcn_tanhf)
#define FAST_TANH(x) __builtin_amdgcn_tanhf(x)
#endif
#endif
#ifndef FAST_TANH
#define FAST_TANH(x) tanhf(x)
#endif

// ---------------------------------------------------------------------------
// CDNA5 WMMA 16x16x32 f16 fragment helpers (wave32).
// A (16x32, f16): lane L holds row m=L&15; half-slot j maps to
//   k = (j<8 ? j : j+8) + 8*(L>>4)  -> two contiguous 8-half (16B) chunks at
//   kbase + 8*(L>>4) and kbase + 16 + 8*(L>>4).
// B (32x16, f16): lane L holds col n=L&15; half-slot j -> k = (j<8 ? j : j+8) + 8*(L>>4).
//   Weights are pre-packed per tile so each lane's fragment is 32 contiguous B.
// C/D (16x16, f32): VGPR r -> row r + 8*(L>>4), col L&15.
// ---------------------------------------------------------------------------
__device__ __forceinline__ v16h frag_a(const _Float16* base, int rowbase, int ldk,
                                       int kbase, int lane) {
  const _Float16* p = base + (rowbase + (lane & 15)) * ldk + kbase + ((lane >> 4) << 3);
  v16h a;
  ((uint4*)&a)[0] = *(const uint4*)(p);
  ((uint4*)&a)[1] = *(const uint4*)(p + 16);
  return a;
}

__device__ __forceinline__ v16h frag_b(const _Float16* packed, int tile, int lane) {
  const _Float16* p = packed + (tile << 9) + (lane << 4);  // 512 halves per tile
  v16h b;
  ((uint4*)&b)[0] = ((const uint4*)p)[0];
  ((uint4*)&b)[1] = ((const uint4*)p)[1];
  return b;
}

__device__ __forceinline__ v8f wmma_tile(const _Float16* As, int rowbase, int ldk,
                                         const _Float16* Bp, int nt, int numKt,
                                         int lane) {
  v8f c = {};
  for (int kt = 0; kt < numKt; ++kt) {
    v16h a = frag_a(As, rowbase, ldk, kt * 32, lane);
    v16h b = frag_b(Bp, nt * numKt + kt, lane);
    c = __builtin_amdgcn_wmma_f32_16x16x32_f16(false, a, false, b, (short)0, c,
                                               false, false);
  }
  return c;
}

__device__ __forceinline__ void store_c_lds(float* out, int rowbase, int ldn,
                                            int nbase, v8f c, int lane) {
  int rofs = rowbase + ((lane >> 4) << 3);
  int n = nbase + (lane & 15);
#pragma unroll
  for (int r = 0; r < 8; ++r) out[(rofs + r) * ldn + n] = c[r];
}

__device__ __forceinline__ float sigmoidf_(float x) {
  return 1.0f / (1.0f + __expf(-x));
}

// ---------------------------------------------------------------------------
// Prologue: repack a row-major f32 weight (N x K view at src[n*ld + coloff + k])
// into f16 WMMA B-fragment tile order: tile(nt,kt) -> 32 lanes x 16 halves.
// ---------------------------------------------------------------------------
__global__ void pack_wmma_b(const float* __restrict__ src, int ld, int coloff,
                            int N, int K, _Float16* __restrict__ dst) {
  int idx = blockIdx.x * blockDim.x + threadIdx.x;
  if (idx >= N * K) return;
  int numKt = K >> 5;
  int tile = idx >> 9;
  int rem = idx & 511;
  int l = rem >> 4;   // lane
  int j = rem & 15;   // half slot
  int nt = tile / numKt, kt = tile % numKt;
  int n = nt * 16 + (l & 15);
  int k = kt * 32 + (j < 8 ? j : j + 8) + ((l >> 4) << 3);
  dst[idx] = (_Float16)src[n * ld + coloff + k];
}

// ---------------------------------------------------------------------------
// Persistent scan kernel: one workgroup per decoder, full T-step scan.
// ---------------------------------------------------------------------------
struct ScanArgs {
  const float* x;                 // (B,T,F)
  const float* Wih0[2];           // (384,3) f32
  const float* bih0[2]; const float* bhh0[2];
  const float* bih1[2]; const float* bhh1[2];
  const _Float16* pWhh0[2]; const _Float16* pWih1[2]; const _Float16* pWhh1[2];
  const _Float16* pWs; const _Float16* pWh; const _Float16* pFc2;
  const float* vW; const float* fc2b;
  float* hist[2];                 // (64,T,128) f32
  float* histWh[2];               // (64,T,128) f32
  float* o[2];                    // (T,B,128) f32 last-layer raw outputs
};

__global__ __launch_bounds__(NTHR, 1) void gru_attn_scan(ScanArgs A) {
  const int d = blockIdx.x;
  const int tid = threadIdx.x;
  const int wv = tid >> 5;
  const int ln = tid & 31;

  const float* Wih0 = A.Wih0[d];
  const float* bih0 = A.bih0[d]; const float* bhh0 = A.bhh0[d];
  const float* bih1 = A.bih1[d]; const float* bhh1 = A.bhh1[d];
  const _Float16* pWhh0 = A.pWhh0[d];
  const _Float16* pWih1 = A.pWih1[d];
  const _Float16* pWhh1 = A.pWhh1[d];
  float* hist = A.hist[d]; float* histWh = A.histWh[d]; float* oOut = A.o[d];

  __shared__ float    hf[ROWS * HID];      // carried h (rows: l*32+b), f32
  __shared__ _Float16 hh[ROWS * HID];      // f16 mirror (WMMA A operand)
  __shared__ float    gx[BATCH * GATES];   // input-side gates
  __shared__ float    gh[BATCH * GATES];   // hidden-side gates
  __shared__ float    sbuf[ROWS * HID];    // s = h @ Ws^T
  __shared__ float    abuf[ROWS * HID];    // logits -> softmax weights
  __shared__ _Float16 cath[ROWS * 256];    // [context | h_raw] f16 (fc2 A)
  __shared__ float    vbuf[HID];
  __shared__ float    xs[BATCH * FIN];

#pragma unroll
  for (int r = 0; r < 8; ++r) {            // zero initial hidden state
    int idx = tid + r * NTHR;
    hf[idx] = 0.0f; hh[idx] = (_Float16)0.0f;
  }
  if (tid < HID) vbuf[tid] = A.vW[tid];
  __syncthreads();

  for (int i = 0; i < TSTEPS; ++i) {
    // stage x_t
    if (tid < BATCH * FIN) {
      int b = tid / FIN, f = tid % FIN;
      xs[tid] = A.x[(b * TSTEPS + i) * FIN + f];
    }
    __syncthreads();

    // ---- layer 0: gx = x @ Wih0^T + bih0 (K=3: VALU) ----
#pragma unroll
    for (int r = 0; r < 12; ++r) {
      int idx = tid + r * NTHR;             // 32*384
      int b = idx / GATES, n = idx % GATES;
      const float* w = Wih0 + n * 3;
      gx[idx] = bih0[n] + xs[b * 3] * w[0] + xs[b * 3 + 1] * w[1] + xs[b * 3 + 2] * w[2];
    }
    __syncthreads();

    // ---- layer 0: gh = h0 @ Whh0^T (WMMA, 48 tiles) ----
    for (int t = wv; t < 48; t += 32) {
      int Mt = t / 24, Nt = t % 24;
      v8f c = wmma_tile(hh, Mt * 16, HID, pWhh0, Nt, 4, ln);
      store_c_lds(gh, Mt * 16, GATES, Nt * 16, c, ln);
    }
    __syncthreads();

    // ---- layer 0 GRU cell ----
#pragma unroll
    for (int r = 0; r < 4; ++r) {
      int idx = tid + r * NTHR;             // 32*128
      int b = idx >> 7, h = idx & 127;
      int g = b * GATES + h;
      float rv = sigmoidf_(gx[g] + gh[g] + bhh0[h]);
      float zv = sigmoidf_(gx[g + HID] + gh[g + HID] + bhh0[HID + h]);
      float nv = FAST_TANH(gx[g + 2 * HID] + rv * (gh[g + 2 * HID] + bhh0[2 * HID + h]));
      float hp = hf[idx];
      float hn = (1.0f - zv) * nv + zv * hp;
      hf[idx] = hn; hh[idx] = (_Float16)hn;
    }
    __syncthreads();

    // ---- layer 1: gx = h0' @ Wih1^T, gh = h1 @ Whh1^T (WMMA, 96 tiles) ----
    for (int t = wv; t < 96; t += 32) {
      int g = t / 48, tt = t % 48;
      int Mt = tt / 24, Nt = tt % 24;
      v8f c = wmma_tile(hh, (g ? BATCH : 0) + Mt * 16, HID,
                        g ? pWhh1 : pWih1, Nt, 4, ln);
      store_c_lds(g ? gh : gx, Mt * 16, GATES, Nt * 16, c, ln);
    }
    __syncthreads();

    // ---- layer 1 GRU cell ----
#pragma unroll
    for (int r = 0; r < 4; ++r) {
      int idx = tid + r * NTHR;
      int b = idx >> 7, h = idx & 127;
      int g = b * GATES + h;
      float rv = sigmoidf_(gx[g] + bih1[h] + gh[g] + bhh1[h]);
      float zv = sigmoidf_(gx[g + HID] + bih1[HID + h] + gh[g + HID] + bhh1[HID + h]);
      float nv = FAST_TANH(gx[g + 2 * HID] + bih1[2 * HID + h] +
                           rv * (gh[g + 2 * HID] + bhh1[2 * HID + h]));
      float hp = hf[BATCH * HID + idx];
      float hn = (1.0f - zv) * nv + zv * hp;
      hf[BATCH * HID + idx] = hn; hh[BATCH * HID + idx] = (_Float16)hn;
    }
    __syncthreads();

    // ---- record hist[i], outputs, cath(h-part) ----
#pragma unroll
    for (int r = 0; r < 8; ++r) {
      int idx = tid + r * NTHR;             // 64*128
      int row = idx >> 7, h = idx & 127;
      float v = hf[idx];
      hist[(row * TSTEPS + i) * HID + h] = v;
      cath[row * 256 + HID + h] = (_Float16)v;
    }
#pragma unroll
    for (int r = 0; r < 4; ++r) {
      int idx = tid + r * NTHR;
      int b = idx >> 7, h = idx & 127;
      oOut[(i * BATCH + b) * HID + h] = hf[BATCH * HID + idx];
    }

    // ---- s = h @ Ws^T (LDS); histWh[i] = h @ Wh^T (global, incremental) ----
    {
      int Mt = wv >> 3, Nt = wv & 7;        // 32 tiles each
      v8f c = wmma_tile(hh, Mt * 16, HID, A.pWs, Nt, 4, ln);
      store_c_lds(sbuf, Mt * 16, HID, Nt * 16, c, ln);
      c = wmma_tile(hh, Mt * 16, HID, A.pWh, Nt, 4, ln);
      int rofs = Mt * 16 + ((ln >> 4) << 3);
      int n = Nt * 16 + (ln & 15);
#pragma unroll
      for (int r = 0; r < 8; ++r)
        histWh[((rofs + r) * TSTEPS + i) * HID + n] = c[r];
    }
    __threadfence();
    __syncthreads();

    // ---- attention scores: logit[row][t] = sum_h tanh(s+histWh)*v ----
    // 16 lanes per (l,b) row; each lane owns 8 fixed h-columns of s and v in
    // registers, streams histWh over t, reduces with 4 shfl_xor steps.
    {
      int row = tid >> 4, sub = tid & 15;
      int hb = sub << 3;
      const float4* spv = (const float4*)(sbuf + row * HID + hb);
      const float4* vpv = (const float4*)(vbuf + hb);
      float4 s0 = spv[0], s1 = spv[1];
      float4 v0 = vpv[0], v1 = vpv[1];
      const float* hwbase = histWh + row * TSTEPS * HID + hb;
      for (int t = 0; t <= i; ++t) {
        const float4* hp = (const float4*)(hwbase + t * HID);
        float4 h0 = hp[0], h1 = hp[1];
        float acc;
        acc  = FAST_TANH(s0.x + h0.x) * v0.x;
        acc += FAST_TANH(s0.y + h0.y) * v0.y;
        acc += FAST_TANH(s0.z + h0.z) * v0.z;
        acc += FAST_TANH(s0.w + h0.w) * v0.w;
        acc += FAST_TANH(s1.x + h1.x) * v1.x;
        acc += FAST_TANH(s1.y + h1.y) * v1.y;
        acc += FAST_TANH(s1.z + h1.z) * v1.z;
        acc += FAST_TANH(s1.w + h1.w) * v1.w;
        acc += __shfl_xor(acc, 1);
        acc += __shfl_xor(acc, 2);
        acc += __shfl_xor(acc, 4);
        acc += __shfl_xor(acc, 8);
        if (sub == 0) abuf[row * HID + t] = acc;
      }
    }
    __syncthreads();

    // ---- softmax over t<=i, one thread per row ----
    if (tid < ROWS) {
      float* ap = abuf + tid * HID;
      float m = ap[0];
      for (int t = 1; t <= i; ++t) m = fmaxf(m, ap[t]);
      float s = 0.0f;
      for (int t = 0; t <= i; ++t) { float e = __expf(ap[t] - m); ap[t] = e; s += e; }
      float inv = 1.0f / s;
      for (int t = 0; t <= i; ++t) ap[t] *= inv;
    }
    __syncthreads();

    // ---- context: c[row][h] = sum_t a[t]*hist[row][t][h] -> cath(c-part) ----
    {
      int row = tid >> 4, hb = (tid & 15) << 3;
      float a0 = 0, a1 = 0, a2 = 0, a3 = 0, a4 = 0, a5 = 0, a6 = 0, a7 = 0;
      const float* hp = hist + row * TSTEPS * HID + hb;
      const float* ap = abuf + row * HID;
      for (int t = 0; t <= i; ++t) {
        float w = ap[t];
        const float4* p = (const float4*)(hp + t * HID);
        float4 u0 = p[0], u1 = p[1];
        a0 += w * u0.x; a1 += w * u0.y; a2 += w * u0.z; a3 += w * u0.w;
        a4 += w * u1.x; a5 += w * u1.y; a6 += w * u1.z; a7 += w * u1.w;
      }
      _Float16* cp = cath + row * 256 + hb;
      cp[0] = (_Float16)a0; cp[1] = (_Float16)a1; cp[2] = (_Float16)a2; cp[3] = (_Float16)a3;
      cp[4] = (_Float16)a4; cp[5] = (_Float16)a5; cp[6] = (_Float16)a6; cp[7] = (_Float16)a7;
    }
    __syncthreads();

    // ---- fc2: att = [c,h] @ fc2^T + b ; carried h <- att (only if i>0) ----
    {
      int Mt = wv >> 3, Nt = wv & 7;        // 32 tiles, K=256 -> 8 k-steps
      v8f c = wmma_tile(cath, Mt * 16, 256, A.pFc2, Nt, 8, ln);
      if (i > 0) {
        int rofs = Mt * 16 + ((ln >> 4) << 3);
        int n = Nt * 16 + (ln & 15);
#pragma unroll
        for (int r = 0; r < 8; ++r) {
          float v = c[r] + A.fc2b[n];
          hf[(rofs + r) * HID + n] = v;
          hh[(rofs + r) * HID + n] = (_Float16)v;
        }
      }
    }
    __syncthreads();
  }
}

// ---------------------------------------------------------------------------
// Output head: out[b,t] = sigmoid(tanh([o1[t,b,:], o2[min(t+D,T-1),b,:]]@W^T+b))
// ---------------------------------------------------------------------------
__global__ void final_out(const float* __restrict__ o1, const float* __restrict__ o2,
                          const float* __restrict__ outW, const float* __restrict__ outb,
                          float* __restrict__ out) {
  int idx = blockIdx.x * blockDim.x + threadIdx.x;
  if (idx >= BATCH * TSTEPS) return;
  int b = idx / TSTEPS, t = idx % TSTEPS;
  int t2 = t + DDELAY; if (t2 > TSTEPS - 1) t2 = TSTEPS - 1;
  const float* p1 = o1 + (t * BATCH + b) * HID;
  const float* p2 = o2 + (t2 * BATCH + b) * HID;
  float acc = outb[0];
#pragma unroll 4
  for (int h = 0; h < HID; ++h) acc += p1[h] * outW[h] + p2[h] * outW[HID + h];
  out[idx] = 1.0f / (1.0f + __expf(-tanhf(acc)));
}

// ---------------------------------------------------------------------------
extern "C" void kernel_launch(void* const* d_in, const int* in_sizes, int n_in,
                              void* d_out, int out_size, void* d_ws, size_t ws_size,
                              hipStream_t stream) {
  (void)in_sizes; (void)n_in; (void)out_size; (void)ws_size;
  const float* attnW = (const float*)d_in[17];
  const float* fc2W  = (const float*)d_in[19];

  char* ws = (char*)d_ws;
  size_t off = 0;
  auto take = [&](size_t bytes) -> char* {
    char* p = ws + off;
    off += (bytes + 255) & ~(size_t)255;
    return p;
  };

  _Float16* pWhh0[2]; _Float16* pWih1[2]; _Float16* pWhh1[2];
  for (int d = 0; d < 2; ++d) {
    pWhh0[d] = (_Float16*)take((size_t)GATES * HID * 2);
    pWih1[d] = (_Float16*)take((size_t)GATES * HID * 2);
    pWhh1[d] = (_Float16*)take((size_t)GATES * HID * 2);
  }
  _Float16* pWs  = (_Float16*)take((size_t)HID * HID * 2);
  _Float16* pWh  = (_Float16*)take((size_t)HID * HID * 2);
  _Float16* pFc2 = (_Float16*)take((size_t)HID * 256 * 2);
  float* hist[2]; float* histWh[2]; float* o[2];
  for (int d = 0; d < 2; ++d) {
    hist[d]   = (float*)take((size_t)ROWS * TSTEPS * HID * 4);
    histWh[d] = (float*)take((size_t)ROWS * TSTEPS * HID * 4);
    o[d]      = (float*)take((size_t)TSTEPS * BATCH * HID * 4);
  }

  // Prologue: repack weights into WMMA fragment order (f16).
  for (int d = 0; d < 2; ++d) {
    const float* Whh0 = (const float*)d_in[2 + d * 8];
    const float* Wih1 = (const float*)d_in[5 + d * 8];
    const float* Whh1 = (const float*)d_in[6 + d * 8];
    pack_wmma_b<<<(GATES * HID + 255) / 256, 256, 0, stream>>>(Whh0, HID, 0, GATES, HID, pWhh0[d]);
    pack_wmma_b<<<(GATES * HID + 255) / 256, 256, 0, stream>>>(Wih1, HID, 0, GATES, HID, pWih1[d]);
    pack_wmma_b<<<(GATES * HID + 255) / 256, 256, 0, stream>>>(Whh1, HID, 0, GATES, HID, pWhh1[d]);
  }
  pack_wmma_b<<<(HID * HID + 255) / 256, 256, 0, stream>>>(attnW, 256, 0,   HID, HID, pWs);
  pack_wmma_b<<<(HID * HID + 255) / 256, 256, 0, stream>>>(attnW, 256, HID, HID, HID, pWh);
  pack_wmma_b<<<(HID * 256 + 255) / 256, 256, 0, stream>>>(fc2W,  256, 0,   HID, 256, pFc2);

  ScanArgs a;
  a.x = (const float*)d_in[0];
  for (int d = 0; d < 2; ++d) {
    a.Wih0[d] = (const float*)d_in[1 + d * 8];
    a.bih0[d] = (const float*)d_in[3 + d * 8];
    a.bhh0[d] = (const float*)d_in[4 + d * 8];
    a.bih1[d] = (const float*)d_in[7 + d * 8];
    a.bhh1[d] = (const float*)d_in[8 + d * 8];
    a.pWhh0[d] = pWhh0[d]; a.pWih1[d] = pWih1[d]; a.pWhh1[d] = pWhh1[d];
    a.hist[d] = hist[d]; a.histWh[d] = histWh[d]; a.o[d] = o[d];
  }
  a.pWs = pWs; a.pWh = pWh; a.pFc2 = pFc2;
  a.vW   = (const float*)d_in[18];
  a.fc2b = (const float*)d_in[20];

  gru_attn_scan<<<2, NTHR, 0, stream>>>(a);

  final_out<<<(BATCH * TSTEPS + 255) / 256, 256, 0, stream>>>(
      o[0], o[1], (const float*)d_in[21], (const float*)d_in[22], (float*)d_out);
}
